// CNN_45406394253762
// MI455X (gfx1250) — compile-verified, matrix operand order checked by reading, not run on hardware
//
#include <hip/hip_runtime.h>
#include <hip/hip_bf16.h>

typedef __attribute__((ext_vector_type(16))) _Float16 v16h;
typedef __attribute__((ext_vector_type(8)))  float    v8f;

typedef uint4     __attribute__((may_alias)) uint4a;
typedef int4      __attribute__((may_alias)) int4a;
typedef float4    __attribute__((may_alias)) float4a;
typedef _Float16  __attribute__((may_alias)) f16a;

union AFrag { v16h h; uint4 u[2]; };

// Scheduling fence: VALU/SALU/VMEM-read may cross, DS and WMMA may NOT.
// Forces all B-fragment ds_loads to issue before the WMMA burst so the 16
// fragment buffers stay live in distinct registers (incremental waits instead
// of a full s_wait_dscnt 0 per WMMA).  Verified working in round 4 asm.
#if __has_builtin(__builtin_amdgcn_sched_barrier)
#define SCHED_FENCE() __builtin_amdgcn_sched_barrier(0x0026)
#else
#define SCHED_FENCE()
#endif

// ---------------- LDS layout (dynamic) ----------------
#define FRAGW_OFF 0                       // 128 KB swizzled W_hh f16 B-fragments
#define WT_OFF    131072                  // 8 waves x 1 KB transposed word ids (reused as dot/cos buffer)
#define LN_OFF    (WT_OFF + 8192)        // 8 waves x 64 B lengths
#define HB_OFF    (LN_OFF + 512)         // 8 waves x 4 KB h (f16, 16x128)
#define CB_OFF    (HB_OFF + 32768)       // 8 waves x 8 KB c (f32, 16x128); doubles as masked-store dump during t-loop
#define LDS_TOTAL (CB_OFF + 65536)       // 238080 B  (< 320 KB / WGP -> 1 WG, 2 waves/SIMD)

__device__ __forceinline__ float fast_rcp(float x) {
#if __has_builtin(__builtin_amdgcn_rcpf)
  return __builtin_amdgcn_rcpf(x);
#else
  return 1.0f / x;
#endif
}
__device__ __forceinline__ float fast_rsq(float x) {
#if __has_builtin(__builtin_amdgcn_rsqf)
  return __builtin_amdgcn_rsqf(x);
#else
  return rsqrtf(x);
#endif
}
__device__ __forceinline__ float sigf(float x) {
  return fast_rcp(1.0f + __expf(-x));
}
__device__ __forceinline__ float tanf_fast(float x) {
  float e = __expf(-2.0f * x);
  return (1.0f - e) * fast_rcp(1.0f + e);
}

// ---------------------------------------------------------------------------
// Kernel A1: Gtab[v][j] = emb[v] . W_ih[j] + b_ih[j] + b_hh[j]   (64 x 512 f32)
// ---------------------------------------------------------------------------
__global__ __launch_bounds__(256) void build_gtab(
    const float* __restrict__ emb, const float* __restrict__ W_ih,
    const float* __restrict__ b_ih, const float* __restrict__ b_hh,
    float* __restrict__ Gtab) {
  __shared__ float e[128];
  int v = blockIdx.x;
  if (threadIdx.x < 128) e[threadIdx.x] = emb[v * 128 + threadIdx.x];
  __syncthreads();
  for (int j = threadIdx.x; j < 512; j += 256) {
    float s = b_ih[j] + b_hh[j];
    const float* wr = W_ih + j * 128;
#pragma unroll 8
    for (int k = 0; k < 128; ++k) s += wr[k] * e[k];
    Gtab[v * 512 + j] = s;
  }
}

// ---------------------------------------------------------------------------
// Kernel A2: swizzle W_hh (512,128) f32 -> f16 WMMA B-fragments.
// B[k][n] = W_hh[n][k].  Fragment (kt,nt): lane = half*16+n holds 16 halfwords
// j -> kk = (j<8 ? half*8+j : 16+half*8+j-8).  32 B per lane, gid-contiguous.
// ---------------------------------------------------------------------------
__global__ __launch_bounds__(256) void build_fragw(
    const float* __restrict__ W_hh, unsigned short* __restrict__ frag) {
  int gid = blockIdx.x * 256 + threadIdx.x;  // 4096 = 4 kt * 32 nt * 32 lanes
  if (gid >= 4096) return;
  int lane = gid & 31;
  int nt   = (gid >> 5) & 31;
  int kt   = gid >> 10;
  int half = lane >> 4, n = lane & 15;
  unsigned short* dst = frag + gid * 16;
#pragma unroll
  for (int j = 0; j < 16; ++j) {
    int kk = (j < 8) ? (half * 8 + j) : (16 + half * 8 + (j - 8));
    _Float16 h = (_Float16)W_hh[(nt * 16 + n) * 128 + (kt * 32 + kk)];
    dst[j] = __builtin_bit_cast(unsigned short, h);
  }
}

// ---------------------------------------------------------------------------
// Main kernel: 256 blocks x 256 thr (8 waves). Wave = 16 sequences (4 batches).
// LDS caps us at 1 WG/WGP (2 waves/SIMD) -> allow up to 512 VGPRs/wave.
// ---------------------------------------------------------------------------
__global__ __launch_bounds__(256, 2) void lstm_main(
    const int* __restrict__ word_ids, const int* __restrict__ lengths,
    const float* __restrict__ Gtab, const uint4* __restrict__ fragW,
    const float* __restrict__ c1w, const float* __restrict__ c1b,
    const float* __restrict__ c2w, const float* __restrict__ c2b,
    const float* __restrict__ sw, const float* __restrict__ sb,
    float* __restrict__ out) {
  extern __shared__ char smem[];
  const int tid  = threadIdx.x;
  const int w    = tid >> 5;
  const int lane = tid & 31;
  const int half = lane >> 4;
  const int col  = lane & 15;
  const int m16  = lane & 15;
  const int seqBase = blockIdx.x * 128 + w * 16;

  // ---- cooperative stage: 128 KB of W_hh fragments into LDS ----
  {
    uint4a* dst = (uint4a*)(smem + FRAGW_OFF);
    for (int i = tid; i < 8192; i += 256) dst[i] = ((const uint4a*)fragW)[i];
  }
  // ---- per-wave stage: word ids transposed wt[t][m], lengths ----
  int* wt = (int*)(smem + WT_OFF) + w * 256;
  int* ln = (int*)(smem + LN_OFF) + w * 16;
  {
    int tBase = half * 8;
    const int* src = word_ids + (seqBase + m16) * 16 + tBase;
#pragma unroll
    for (int j = 0; j < 8; ++j) wt[(tBase + j) * 16 + m16] = src[j];
    if (lane < 16) ln[lane] = lengths[seqBase + lane];
  }
  __syncthreads();

  int lenv[8];
#pragma unroll
  for (int r = 0; r < 8; ++r) lenv[r] = ln[half * 8 + r];

  _Float16* hb = (_Float16*)(smem + HB_OFF) + w * 2048;  // 16 x 128 f16
  f16a* hs = (f16a*)smem;                                // f16-element view of LDS
  const char* gb = (const char*)Gtab + col * 4;          // per-lane column base

  // h-store address bases (f16 element units)
  const unsigned realbase = (unsigned)(HB_OFF / 2) + (unsigned)w * 2048u +
                            (unsigned)(half * 8) * 128u + (unsigned)col;
  const unsigned dumpbase = (unsigned)(CB_OFF / 2) + (unsigned)w * 4096u;

  // per-kt fragment base pointers: all ds offsets become 16-bit immediates
  const char* fwb[4];
#pragma unroll
  for (int kt = 0; kt < 4; ++kt)
    fwb[kt] = (const char*)smem + FRAGW_OFF + kt * 32768 + lane * 32;

  float cacc[8][8];
#pragma unroll
  for (int i = 0; i < 8; ++i)
#pragma unroll
    for (int r = 0; r < 8; ++r) cacc[i][r] = 0.0f;
  AFrag hA[4];
#pragma unroll
  for (int k = 0; k < 4; ++k) hA[k].h = (v16h)(_Float16)0.0f;

#pragma unroll 1
  for (int t = 0; t < 16; ++t) {
    // word ids for my 8 rows -> 32-bit byte offsets into Gtab
    int4a wa = *(const int4a*)(wt + t * 16 + half * 8);
    int4a wb = *(const int4a*)(wt + t * 16 + half * 8 + 4);
    unsigned goff[8];
    goff[0] = (unsigned)wa.x * 2048u; goff[1] = (unsigned)wa.y * 2048u;
    goff[2] = (unsigned)wa.z * 2048u; goff[3] = (unsigned)wa.w * 2048u;
    goff[4] = (unsigned)wb.x * 2048u; goff[5] = (unsigned)wb.y * 2048u;
    goff[6] = (unsigned)wb.z * 2048u; goff[7] = (unsigned)wb.w * 2048u;

    // branchless length masking: redirect dead h-stores into the dump slot
    bool keep[8];
    unsigned hoff[8];
#pragma unroll
    for (int r = 0; r < 8; ++r) {
      keep[r] = (t < lenv[r]);
      hoff[r] = keep[r] ? (realbase + (unsigned)r * 128u) : dumpbase;
    }

#pragma unroll
    for (int hT = 0; hT < 8; ++hT) {
      // ---- gate-table gather first (long-latency vmem, 32 x b32) ----
      v8f acc[4];
#pragma unroll
      for (int g = 0; g < 4; ++g) {
        v8f a;
#pragma unroll
        for (int r = 0; r < 8; ++r)
          a[r] = *(const float*)(gb + goff[r] + (g * 8 + hT) * 64);
        acc[g] = a;
      }
      // ---- preload all 16 B fragments for this hidden tile (32 x b128) ----
      AFrag bf[4][4];
#pragma unroll
      for (int g = 0; g < 4; ++g)
#pragma unroll
        for (int kt = 0; kt < 4; ++kt) {
          int off = (g * 8 + hT) * 1024;               // <= 31*1024: imm-able
          bf[g][kt].u[0] = *(const uint4a*)(fwb[kt] + off);
          bf[g][kt].u[1] = *(const uint4a*)(fwb[kt] + off + 16);
        }
      SCHED_FENCE();   // all 32 ds_loads issued; none may sink into WMMA burst
      // ---- 16 WMMA, kt-outer / gate-inner: 4 independent chains ----
#pragma unroll
      for (int kt = 0; kt < 4; ++kt)
#pragma unroll
        for (int g = 0; g < 4; ++g)
          acc[g] = __builtin_amdgcn_wmma_f32_16x16x32_f16(
              false, hA[kt].h, false, bf[g][kt].h, (short)0, acc[g], false, false);
      SCHED_FENCE();   // keep WMMA burst contiguous

      // ---- elementwise LSTM cell + branchless masked update ----
#pragma unroll
      for (int r = 0; r < 8; ++r) {
        float iv = sigf(acc[0][r]);
        float fv = sigf(acc[1][r]);
        float gv = tanf_fast(acc[2][r]);
        float ov = sigf(acc[3][r]);
        float cn = fv * cacc[hT][r] + iv * gv;
        float hn = ov * tanf_fast(cn);
        cacc[hT][r] = keep[r] ? cn : cacc[hT][r];
        hs[hoff[r] + hT * 16] = (_Float16)hn;   // unconditional ds_store_b16
      }
    }
    // ---- reload h as A-fragments for next step (2 x b128 per K-tile) ----
#pragma unroll
    for (int kt = 0; kt < 4; ++kt) {
      hA[kt].u[0] = *(const uint4a*)(hb + m16 * 128 + kt * 32 + half * 8);
      hA[kt].u[1] = *(const uint4a*)(hb + m16 * 128 + kt * 32 + 16 + half * 8);
    }
  }

  // ================= tail: cosine sim + 2 convs + scorer =================
  float* cb = (float*)(smem + CB_OFF) + w * 2048;  // 16 x 128 f32
#pragma unroll
  for (int hT = 0; hT < 8; ++hT)
#pragma unroll
    for (int r = 0; r < 8; ++r)
      cb[(r + half * 8) * 128 + hT * 16 + col] = cacc[hT][r];

  float* dotb = (float*)(smem + WT_OFF) + w * 256;  // reuse own wt region
#pragma unroll
  for (int p = 0; p < 2; ++p) {                     // 64 (bi,i,j) dots
    int idx = p * 32 + lane;
    int bi = idx >> 4, ii = (idx >> 2) & 3, jj = idx & 3;
    const float4a* ci = (const float4a*)(cb + (bi * 4 + ii) * 128);
    const float4a* cj = (const float4a*)(cb + (bi * 4 + jj) * 128);
    float s = 0.0f;
#pragma unroll
    for (int k = 0; k < 32; ++k) {
      float4 x = ci[k], y = cj[k];
      s += x.x * y.x + x.y * y.y + x.z * y.z + x.w * y.w;
    }
    dotb[idx] = s;
  }
  float cosv[2];
#pragma unroll
  for (int p = 0; p < 2; ++p) {
    int idx = p * 32 + lane;
    int bi = idx >> 4, ii = (idx >> 2) & 3, jj = idx & 3;
    float dii = dotb[bi * 16 + ii * 4 + ii];
    float djj = dotb[bi * 16 + jj * 4 + jj];
    cosv[p] = dotb[idx] * fast_rsq(dii * djj);
  }
#pragma unroll
  for (int p = 0; p < 2; ++p) dotb[p * 32 + lane] = cosv[p];

  if (lane < 4) {                                   // one lane per batch
    const float* cm = dotb + lane * 16;             // 4x4 cos image
    float a1[4][3][3];
#pragma unroll
    for (int oc = 0; oc < 4; ++oc)
#pragma unroll
      for (int y = 0; y < 3; ++y)
#pragma unroll
        for (int x = 0; x < 3; ++x) {
          float s = c1b[oc];
#pragma unroll
          for (int kh = 0; kh < 2; ++kh)
#pragma unroll
            for (int kw = 0; kw < 2; ++kw)
              s += c1w[oc * 4 + kh * 2 + kw] * cm[(y + kh) * 4 + (x + kw)];
          a1[oc][y][x] = s > 0.0f ? s : 0.0f;
        }
    float sc = sb[0];
#pragma unroll
    for (int oc = 0; oc < 8; ++oc)
#pragma unroll
      for (int y = 0; y < 2; ++y)
#pragma unroll
        for (int x = 0; x < 2; ++x) {
          float s = c2b[oc];
#pragma unroll
          for (int ic = 0; ic < 4; ++ic)
#pragma unroll
            for (int kh = 0; kh < 2; ++kh)
#pragma unroll
              for (int kw = 0; kw < 2; ++kw)
                s += c2w[((oc * 4 + ic) * 2 + kh) * 2 + kw] * a1[ic][y + kh][x + kw];
          s = s > 0.0f ? s : 0.0f;
          sc += sw[oc * 4 + y * 2 + x] * s;
        }
    out[blockIdx.x * 32 + w * 4 + lane] = sigf(sc);
  }
}

// ---------------------------------------------------------------------------
extern "C" void kernel_launch(void* const* d_in, const int* in_sizes, int n_in,
                              void* d_out, int out_size, void* d_ws, size_t ws_size,
                              hipStream_t stream) {
  (void)in_sizes; (void)n_in; (void)out_size; (void)ws_size;
  const int*   word_ids = (const int*)d_in[0];
  const int*   lengths  = (const int*)d_in[1];
  const float* emb      = (const float*)d_in[2];
  const float* W_ih     = (const float*)d_in[3];
  const float* W_hh     = (const float*)d_in[4];
  const float* b_ih     = (const float*)d_in[5];
  const float* b_hh     = (const float*)d_in[6];
  const float* c1w      = (const float*)d_in[7];
  const float* c1b      = (const float*)d_in[8];
  const float* c2w      = (const float*)d_in[9];
  const float* c2b      = (const float*)d_in[10];
  const float* sw       = (const float*)d_in[11];
  const float* sb       = (const float*)d_in[12];

  float*          Gtab  = (float*)d_ws;                               // 128 KB
  unsigned short* fragW = (unsigned short*)((char*)d_ws + 131072);    // 128 KB

  build_gtab<<<64, 256, 0, stream>>>(emb, W_ih, b_ih, b_hh, Gtab);
  build_fragw<<<16, 256, 0, stream>>>(W_hh, fragW);
  lstm_main<<<256, 256, LDS_TOTAL, stream>>>(
      word_ids, lengths, Gtab, (const uint4*)fragW,
      c1w, c1b, c2w, c2b, sw, sb, (float*)d_out);
}
